// MyConv2d_7395933683730
// MI455X (gfx1250) — matrix operand order repaired and checked by reading, not compiled, not run on hardware
//
#include <hip/hip_runtime.h>

typedef _Float16 h16;
typedef __attribute__((ext_vector_type(16))) _Float16 v16h;
typedef __attribute__((ext_vector_type(8)))  _Float16 v8h;
typedef __attribute__((ext_vector_type(8)))  float    v8f;

#define Bv   32
#define Cv   256
#define Ov   256
#define Hv   56
#define Wv   56
#define HWv  (Hv*Wv)          // 3136
#define TM   64               // out-channels per block (4 subtiles of 16)
#define TR   4                // output rows per block
#define CK   32               // input-channel chunk (WMMA K)
#define XROWS (TR+2)          // 6 (halo rows)
#define XCOLS (Wv+2)          // 58 (halo cols)
#define XPIX  (XROWS*XCOLS)   // 348 pixels
#define XPLANE (XPIX*8)       // 2784 elems per 8-channel plane
#define WROWS (9*TM)          // 576 weight rows
#define WPLANE (WROWS*8)      // 4608 elems per 8-channel plane

// 16-value f16 fragment: two 16B LDS reads from planes p and p+stride elems.
static __device__ __forceinline__ v16h load16(const h16* p, int stride) {
    v8h lo = *(const v8h*)(p);
    v8h hi = *(const v8h*)(p + stride);
    v16h f;
#pragma unroll
    for (int e = 0; e < 8; ++e) { f[e] = lo[e]; f[e + 8] = hi[e]; }
    return f;
}

// 256 threads, >=4 waves/SIMD resident -> <=128 VGPRs -> 2 workgroups per WGP
// so one block's WMMA phase overlaps the other's LDS staging phase.
__global__ __launch_bounds__(256, 4)
void conv3x3_wmma_f16(const float* __restrict__ x,
                      const float* __restrict__ wt,
                      float* __restrict__ out)
{
    // Bank-conflict-free layouts: [c>>3 plane][pixel or row][c&7]
    __shared__ h16 xs[4 * XPLANE];   // 22272 B
    __shared__ h16 ws[4 * WPLANE];   // 36864 B

    const int tid  = threadIdx.x;
    const int lane = tid & 31;
    const int wv   = tid >> 5;      // wave 0..7
    const int mi   = wv & 3;        // M subtile
    const int nh   = wv >> 2;       // N half (0/1)
    const int ln   = lane & 15;
    const int khi  = lane >> 4;     // K-half select per WMMA layout

    const int hb = blockIdx.x % 14;
    const int b  = blockIdx.x / 14;
    const int h0 = hb * TR;
    const int o0 = blockIdx.y * TM;

    // B fragment per-lane base: plane 2*khi at the lane's pixel (tap adds (kh*58+kw)*8)
    int boff[7];
#pragma unroll
    for (int j = 0; j < 7; ++j) {
        int idx = (nh * 7 + j) * 16 + ln;        // 0..223 flat spatial in tile
        int r   = idx / Wv;
        int cw  = idx - r * Wv;
        boff[j] = 2 * khi * XPLANE + (r * XCOLS + cw) * 8;
    }
    // A fragment: planes 2*khi and 2*khi+1 at row mi*16+ln (tap adds 64*8 = 512)
    const int aoff = 2 * khi * WPLANE + (mi * 16 + ln) * 8;

    v8f acc[7] = {};

    // One-time clear: halo pixels stay zero for the whole K loop.
    for (int i = tid; i < 4 * XPLANE; i += 256) xs[i] = (h16)0.f;
    __syncthreads();

    for (int c0 = 0; c0 < Cv; c0 += CK) {
        // ---- stage x interior: float4 rows (halo already zero) ----
        for (int i4 = tid; i4 < CK * XROWS * 14; i4 += 256) {   // 2688 float4s
            int c   = i4 / (XROWS * 14);
            int rem = i4 - c * (XROWS * 14);
            int r   = rem / 14;
            int q   = rem - r * 14;
            int hin = h0 - 1 + r;
            if ((unsigned)hin < (unsigned)Hv) {
                const float4* src =
                    (const float4*)(x + (((size_t)b * Cv + c0 + c) * Hv + hin) * Wv) + q;
                float4 v = *src;
                int base = ((c >> 3) * XPIX + r * XCOLS + 4 * q + 1) * 8 + (c & 7);
                xs[base     ] = (h16)v.x;
                xs[base +  8] = (h16)v.y;
                xs[base + 16] = (h16)v.z;
                xs[base + 24] = (h16)v.w;
            }
        }
        // ---- stage weights: [c>>3][tap*64+o][c&7], float4 over contiguous [c][tap] ----
        for (int i4 = tid; i4 < TM * 72; i4 += 256) {           // 4608 float4s, 18 iters
            int o     = i4 / 72;
            int ebase = (i4 - o * 72) * 4;                      // element within 288
            const float4* src =
                (const float4*)(wt + (size_t)(o0 + o) * (Cv * 9) + (size_t)c0 * 9) + (i4 - o * 72);
            float4 v = *src;
            float vv[4] = {v.x, v.y, v.z, v.w};
#pragma unroll
            for (int q = 0; q < 4; ++q) {
                int e   = ebase + q;
                int c   = e / 9;
                int tap = e - c * 9;
                ws[((c >> 3) * WROWS + tap * TM + o) * 8 + (c & 7)] = (h16)vv[q];
            }
        }
        __syncthreads();

        // ---- pipelined compute: prefetch next fragment while WMMA runs ----
        v16h Acur = load16(&ws[aoff], WPLANE);
        v16h Bcur = load16(&xs[boff[0]], XPLANE);
        v16h Anx, Bnx;
#pragma unroll
        for (int t = 0; t < 9; ++t) {
            const int kh = t / 3, kw = t - kh * 3;
            const int toff = (kh * XCOLS + kw) * 8;
#pragma unroll
            for (int j = 0; j < 7; ++j) {
                if (j == 0 && t < 8)
                    Anx = load16(&ws[aoff + (t + 1) * (TM * 8)], WPLANE);
                if (j < 6) {
                    Bnx = load16(&xs[boff[j + 1] + toff], XPLANE);
                } else if (t < 8) {
                    const int kh2 = (t + 1) / 3, kw2 = (t + 1) - kh2 * 3;
                    Bnx = load16(&xs[boff[0] + (kh2 * XCOLS + kw2) * 8], XPLANE);
                }
                acc[j] = __builtin_amdgcn_wmma_f32_16x16x32_f16(
                    false, Acur, false, Bcur, (short)0, acc[j], false, false);
                if (j < 6 || t < 8) Bcur = Bnx;
            }
            if (t < 8) Acur = Anx;
        }
        __syncthreads();
    }

    // ---- epilogue: out flat layout is [O][B][H][W] ----
#pragma unroll
    for (int j = 0; j < 7; ++j) {
        int idx = (nh * 7 + j) * 16 + ln;
        int r   = idx / Wv;
        int cw  = idx - r * Wv;
        int ob  = o0 + mi * 16 + khi * 8;
#pragma unroll
        for (int v = 0; v < 8; ++v) {
            size_t o = (size_t)(ob + v);
            out[(o * Bv + b) * HWv + (size_t)(h0 + r) * Wv + cw] = acc[j][v];
        }
    }
}

extern "C" void kernel_launch(void* const* d_in, const int* in_sizes, int n_in,
                              void* d_out, int out_size, void* d_ws, size_t ws_size,
                              hipStream_t stream) {
    (void)in_sizes; (void)n_in; (void)d_ws; (void)ws_size; (void)out_size;
    const float* x  = (const float*)d_in[0];
    const float* wt = (const float*)d_in[1];
    float* out      = (float*)d_out;
    dim3 grid(14 * Bv, Ov / TM);   // 448 x 4
    conv3x3_wmma_f16<<<grid, 256, 0, stream>>>(x, wt, out);
}
